// GptDecoderLayer_26568667693450
// MI455X (gfx1250) — matrix operand order repaired and checked by reading, not compiled, not run on hardware
//
#include <hip/hip_runtime.h>

#define S_LEN   2048
#define D_MODEL 2048
#define NH      16
#define KVH     4
#define HD      128
#define FFN     5632

typedef __attribute__((ext_vector_type(16))) __bf16 v16bf;
typedef __attribute__((ext_vector_type(8)))  float  v8f;
typedef __attribute__((ext_vector_type(2)))  __bf16 v2bf;

union FragAB { v16bf v; unsigned int u[8]; };
union FragC  { v8f   v; float        f[8]; };

__device__ __forceinline__ unsigned short f2bf(float f) {
  union { float f; unsigned int u; } x; x.f = f;
  unsigned int r = x.u + 0x7FFFu + ((x.u >> 16) & 1u);  // round-to-nearest-even
  return (unsigned short)(r >> 16);
}

__device__ __forceinline__ unsigned int pack2bf(float lo, float hi) {
#if __has_builtin(__builtin_amdgcn_cvt_pk_bf16_f32)
  union { v2bf v; unsigned int u; } p;
  p.v = __builtin_amdgcn_cvt_pk_bf16_f32(lo, hi);
  return p.u;
#else
  return (unsigned int)f2bf(lo) | ((unsigned int)f2bf(hi) << 16);
#endif
}

// async copy 16B global -> LDS (ASYNCcnt), per-lane addresses
__device__ __forceinline__ void async_b128(unsigned lds_byte_addr, const void* gaddr) {
  unsigned long long ga = (unsigned long long)(uintptr_t)gaddr;
  asm volatile("global_load_async_to_lds_b128 %0, %1, off"
               :: "v"(lds_byte_addr), "v"(ga) : "memory");
}
__device__ __forceinline__ void wait_async0() {
  asm volatile("s_wait_asynccnt 0x0" ::: "memory");
}

// ---------------------------------------------------------------- RMSNorm ----
__global__ __launch_bounds__(256)
void k_rmsnorm_bf16(const float* __restrict__ x, const float* __restrict__ scale,
                    unsigned short* __restrict__ out) {
  __shared__ float red[256];
  int row = blockIdx.x;
  const float* xr = x + (size_t)row * D_MODEL;
  float ss = 0.f;
  for (int c = threadIdx.x; c < D_MODEL; c += 256) { float v = xr[c]; ss += v * v; }
  red[threadIdx.x] = ss;
  __syncthreads();
  for (int s = 128; s > 0; s >>= 1) {
    if (threadIdx.x < s) red[threadIdx.x] += red[threadIdx.x + s];
    __syncthreads();
  }
  float rs = rsqrtf(red[0] / (float)D_MODEL + 1e-6f);
  for (int c = threadIdx.x; c < D_MODEL; c += 256)
    out[(size_t)row * D_MODEL + c] = f2bf(xr[c] * rs * scale[c]);
}

// ------------------------------------------------------------ WMMA GEMM ------
// C[M,N](f32) = A[M,K](bf16) @ B[K,N](f32->bf16) (+ residual)
// block tile 128x128, 256 threads (8 waves); wave = 32 rows x 64 cols
// (8 WMMA per 32-K step). Double-buffered LDS, A tile via async-to-LDS,
// B tile fp32->bf16 packed. Requires K % 64 == 0 (nk even).
__global__ __launch_bounds__(256)
void k_gemm_bf16(const unsigned short* __restrict__ A, const float* __restrict__ B,
                 float* __restrict__ C, const float* __restrict__ resid,
                 int M, int N, int K) {
  __shared__ unsigned int a_lds[2][128 * 20];  // 128 rows x 32 bf16; stride 20 dw
  __shared__ unsigned int b_lds[2][128 * 20];  // 128 cols x 32 bf16 (K-transposed)

  int tid  = threadIdx.x;
  int lane = tid & 31, wid = tid >> 5;
  int rw = wid & 3, cw = wid >> 2;             // 4 row-groups(32) x 2 col-groups(64)
  int hi = lane >> 4, l15 = lane & 15;
  int cm = blockIdx.y * 128, cn = blockIdx.x * 128;
  const unsigned int* A32 = (const unsigned int*)A;
  unsigned uN = (unsigned)N, uK = (unsigned)K;

  // A tile: 128x32 bf16 = 512 b128 chunks -> 2 async b128 per thread per stage
  int ar = tid >> 2, acu = (tid & 3) * 4;       // rows ar and ar+64
  const unsigned int* aNext = A32 + ((unsigned)(cm + ar) * (uK >> 1) + (unsigned)acu);
  unsigned aoff2   = 32u * uK;                  // +64 rows in dwords
  unsigned a_laddr = (unsigned)(uintptr_t)&a_lds[0][ar * 20 + acu];

  // B tile: 32x128 fp32; each thread packs 8 kr-pairs (2 rows -> 1 dword)
  const float* bNext = B + (unsigned)cn;

  FragC acc[2][4];
#pragma unroll
  for (int s = 0; s < 2; s++)
#pragma unroll
    for (int t = 0; t < 4; t++)
#pragma unroll
      for (int i = 0; i < 8; i++) acc[s][t].f[i] = 0.f;

  auto stage = [&](int buf) {
    unsigned lbase = a_laddr + (unsigned)buf * (128 * 20 * 4);
    async_b128(lbase, aNext);
    async_b128(lbase + 64 * 20 * 4, aNext + aoff2);
#pragma unroll
    for (int i = 0; i < 8; i++) {
      int linear = tid + 256 * i;
      int p = linear >> 7, n = linear & 127;            // kr pair p, col n
      unsigned gi = (unsigned)(2 * p) * uN + (unsigned)n;
      float f0 = bNext[gi];
      float f1 = bNext[gi + uN];
      b_lds[buf][(unsigned)n * 20 + (unsigned)p] = pack2bf(f0, f1);
    }
    aNext += 16;                 // +32 bf16 along K
    bNext += (size_t)32 * uN;    // +32 K-rows
  };

  auto compute = [&](int buf) {
    FragAB a0, a1;
#pragma unroll
    for (int v = 0; v < 8; v++) {
      int kk = (v < 4 ? 2 * v : 16 + 2 * (v - 4)) + hi * 8;   // ISA A-layout
      a0.u[v] = a_lds[buf][(rw * 32 + l15) * 20 + (kk >> 1)];
      a1.u[v] = a_lds[buf][(rw * 32 + 16 + l15) * 20 + (kk >> 1)];
    }
#pragma unroll
    for (int t = 0; t < 4; t++) {
      FragAB b;
#pragma unroll
      for (int v = 0; v < 8; v++) {
        int kk = (v < 4 ? 2 * v : 16 + 2 * (v - 4)) + hi * 8;   // same k order as A
        b.u[v] = b_lds[buf][(cw * 64 + t * 16 + l15) * 20 + (kk >> 1)];
      }
      acc[0][t].v = __builtin_amdgcn_wmma_f32_16x16x32_bf16(false, a0.v, false, b.v,
                                                            (short)0, acc[0][t].v, false, false);
      acc[1][t].v = __builtin_amdgcn_wmma_f32_16x16x32_bf16(false, a1.v, false, b.v,
                                                            (short)0, acc[1][t].v, false, false);
    }
  };

  int nk = K >> 5;               // even for all our K (multiples of 64)
  stage(0);
  wait_async0();
  __syncthreads();

  for (int ks = 0; ks < nk; ks += 2) {
    if (ks + 1 < nk) stage(1);
    compute(0);
    wait_async0();
    __syncthreads();
    if (ks + 2 < nk) stage(0);
    compute(1);
    wait_async0();
    __syncthreads();
  }

  // epilogue: branch ONCE on residual, clean store clauses
  if (resid != nullptr) {
#pragma unroll
    for (int s = 0; s < 2; s++)
#pragma unroll
      for (int v = 0; v < 8; v++) {
        unsigned rowoff = (unsigned)(cm + rw * 32 + s * 16 + v + hi * 8) * uN +
                          (unsigned)(cn + cw * 64 + l15);
#pragma unroll
        for (int t = 0; t < 4; t++)
          C[rowoff + t * 16] = acc[s][t].f[v] + resid[rowoff + t * 16];
      }
  } else {
#pragma unroll
    for (int s = 0; s < 2; s++)
#pragma unroll
      for (int v = 0; v < 8; v++) {
        unsigned rowoff = (unsigned)(cm + rw * 32 + s * 16 + v + hi * 8) * uN +
                          (unsigned)(cn + cw * 64 + l15);
#pragma unroll
        for (int t = 0; t < 4; t++)
          C[rowoff + t * 16] = acc[s][t].f[v];
      }
  }
}

// ---------------------------------------------------------------- RoPE -------
// grid (nh*64/256, S), block 256
__global__ void k_rope_bf16(const float* __restrict__ in,
                            unsigned short* __restrict__ out, int nh) {
  int idx = blockIdx.x * 256 + threadIdx.x;   // within row: nh*64
  int i = idx & 63;
  int h = idx >> 6;
  int s = blockIdx.y;
  unsigned base = (unsigned)s * (unsigned)(nh * HD) + (unsigned)h * HD + (unsigned)i;
  float x1 = in[base], x2 = in[base + 64];
  float freq = (float)s * __expf(-(float)i * (9.210340371976184f / 64.f)); // 10000^(-i/64)
  float c = __cosf(freq), sn = __sinf(freq);
  out[base]      = f2bf(x1 * c - x2 * sn);
  out[base + 64] = f2bf(x2 * c + x1 * sn);
}

__global__ void k_f32_to_bf16(const float* __restrict__ in,
                              unsigned short* __restrict__ out, int n) {
  int i = blockIdx.x * blockDim.x + threadIdx.x;
  if (i < n) out[i] = f2bf(in[i]);
}

// ---------------------------------------------------- flash attention --------
// grid (S/64, NH), 128 threads (4 waves); wave = 16 queries; 32-key tiles.
__global__ __launch_bounds__(128)
void k_flash_attn(const unsigned short* __restrict__ qb,
                  const unsigned short* __restrict__ kb,
                  const unsigned short* __restrict__ vb,
                  unsigned short* __restrict__ ob) {
  __shared__ unsigned int k_lds[32 * 68];    // [key][128 dims]; stride 68 dw
  __shared__ unsigned int v_lds[128 * 20];   // [dim][32 keys];  stride 20 dw
  __shared__ unsigned int p_lds[4 * 16 * 20];
  unsigned short* v16 = (unsigned short*)v_lds;
  unsigned short* p16 = (unsigned short*)p_lds;

  int tid = threadIdx.x, lane = tid & 31, wid = tid >> 5;
  int hi = lane >> 4, l15 = lane & 15;
  int h = blockIdx.y, kvh = h >> 2;                 // n_rep = 4
  int qbase = blockIdx.x * 64 + wid * 16;
  const unsigned int* q32 = (const unsigned int*)qb;
  const unsigned int* k32 = (const unsigned int*)kb;
  const unsigned int* v32 = (const unsigned int*)vb;

  // K-tile async copy setup: 2048 dw = 512 b128 chunks, 4 per thread
  unsigned k_goff[4], k_laddr[4];
#pragma unroll
  for (int i = 0; i < 4; i++) {
    int c = tid + 128 * i;
    int kr = c >> 4, cu4 = (c & 15) * 4;
    k_goff[i]  = (unsigned)kr * (KVH * HD / 2) + (unsigned)(kvh * 64 + cu4);
    k_laddr[i] = (unsigned)(uintptr_t)&k_lds[kr * 68 + cu4];
  }

  // Q fragments: 16x128 bf16 = 4 K-steps
  FragAB qf[4];
  unsigned qrow = (unsigned)(qbase + l15);
#pragma unroll
  for (int s = 0; s < 4; s++)
#pragma unroll
    for (int v = 0; v < 8; v++) {
      int kk = (v < 4 ? 2 * v : 16 + 2 * (v - 4)) + hi * 8 + 32 * s;
      qf[s].u[v] = q32[qrow * (D_MODEL / 2) + (unsigned)(h * (HD / 2)) + (unsigned)(kk >> 1)];
    }

  FragC o[8];
#pragma unroll
  for (int t = 0; t < 8; t++)
#pragma unroll
    for (int i = 0; i < 8; i++) o[t].f[i] = 0.f;
  float mst[8], lst[8];
#pragma unroll
  for (int r = 0; r < 8; r++) { mst[r] = -3.0e38f; lst[r] = 0.f; }

  const float scl = 0.08838834764831845f;           // 1/sqrt(128)
  int nkb = blockIdx.x * 2 + 2;                     // uniform across block
  int wqmax = qbase + 15;

  for (int kbi = 0; kbi < nkb; kbi++) {
    unsigned key0 = (unsigned)kbi * 32u;
    unsigned kadv = key0 * (KVH * HD / 2);          // dword advance for this tile
    // async stage K tile [32 keys][128 dims]
#pragma unroll
    for (int i = 0; i < 4; i++)
      async_b128(k_laddr[i], k32 + (kadv + k_goff[i]));
    // sync stage V tile transposed [128 dims][32 keys]
#pragma unroll
    for (int i = 0; i < 16; i++) {
      int linear = tid + 128 * i;
      int kr = linear >> 6, cu = linear & 63;
      unsigned int w = v32[kadv + (unsigned)kr * (KVH * HD / 2) +
                           (unsigned)(kvh * 64 + cu)];
      int d = cu * 2;
      v16[d * 40 + kr]       = (unsigned short)(w & 0xFFFFu);
      v16[(d + 1) * 40 + kr] = (unsigned short)(w >> 16);
    }
    wait_async0();
    __syncthreads();

    if ((int)key0 <= wqmax) {                       // wave-uniform causal skip
      FragC s1, s2;
#pragma unroll
      for (int i = 0; i < 8; i++) { s1.f[i] = 0.f; s2.f[i] = 0.f; }
#pragma unroll
      for (int s = 0; s < 4; s++) {
        FragAB b0, b1;
#pragma unroll
        for (int v = 0; v < 8; v++) {
          int kk = (v < 4 ? 2 * v : 16 + 2 * (v - 4)) + hi * 8;
          b0.u[v] = k_lds[l15 * 68        + ((32 * s + kk) >> 1)];
          b1.u[v] = k_lds[(16 + l15) * 68 + ((32 * s + kk) >> 1)];
        }
        s1.v = __builtin_amdgcn_wmma_f32_16x16x32_bf16(false, qf[s].v, false, b0.v,
                                                       (short)0, s1.v, false, false);
        s2.v = __builtin_amdgcn_wmma_f32_16x16x32_bf16(false, qf[s].v, false, b1.v,
                                                       (short)0, s2.v, false, false);
      }
      // online softmax: rows live one-per-VGPR across 16-lane halves
#pragma unroll
      for (int r = 0; r < 8; r++) {
        int qr = qbase + r + hi * 8;
        float a = s1.f[r] * scl;
        float b = s2.f[r] * scl;
        if ((int)key0 + l15 > qr)      a = -1e30f;
        if ((int)key0 + 16 + l15 > qr) b = -1e30f;
        float mx = fmaxf(a, b);
        mx = fmaxf(mx, __shfl_xor(mx, 1, 32));
        mx = fmaxf(mx, __shfl_xor(mx, 2, 32));
        mx = fmaxf(mx, __shfl_xor(mx, 4, 32));
        mx = fmaxf(mx, __shfl_xor(mx, 8, 32));
        float mnew  = fmaxf(mst[r], mx);
        float alpha = __expf(mst[r] - mnew);
        float p1 = __expf(a - mnew);
        float p2 = __expf(b - mnew);
        float rs = p1 + p2;
        rs += __shfl_xor(rs, 1, 32);
        rs += __shfl_xor(rs, 2, 32);
        rs += __shfl_xor(rs, 4, 32);
        rs += __shfl_xor(rs, 8, 32);
        lst[r] = lst[r] * alpha + rs;
        mst[r] = mnew;
#pragma unroll
        for (int t = 0; t < 8; t++) o[t].f[r] *= alpha;
        int prow = r + hi * 8;
        p16[wid * 16 * 40 + prow * 40 + l15]      = f2bf(p1);
        p16[wid * 16 * 40 + prow * 40 + 16 + l15] = f2bf(p2);
      }
      // re-lay P into A-fragment (wave-private LDS round trip)
      FragAB pf;
#pragma unroll
      for (int v = 0; v < 8; v++) {
        int kk = (v < 4 ? 2 * v : 16 + 2 * (v - 4)) + hi * 8;
        pf.u[v] = p_lds[wid * 16 * 20 + l15 * 20 + (kk >> 1)];
      }
      // O += P @ V over 8 column tiles
#pragma unroll
      for (int t = 0; t < 8; t++) {
        FragAB bv;
#pragma unroll
        for (int v = 0; v < 8; v++) {
          int kk = (v < 4 ? 2 * v : 16 + 2 * (v - 4)) + hi * 8;
          bv.u[v] = v_lds[(t * 16 + l15) * 20 + (kk >> 1)];
        }
        o[t].v = __builtin_amdgcn_wmma_f32_16x16x32_bf16(false, pf.v, false, bv.v,
                                                         (short)0, o[t].v, false, false);
      }
    }
    __syncthreads();
  }

  // normalize + store O (bf16, [S, H*HD])
#pragma unroll
  for (int t = 0; t < 8; t++)
#pragma unroll
    for (int r = 0; r < 8; r++) {
      unsigned row = (unsigned)(qbase + r + hi * 8);
      float val = o[t].f[r] / lst[r];
      ob[row * (unsigned)D_MODEL + (unsigned)(h * HD + t * 16 + l15)] = f2bf(val);
    }
}

// --------------------------------------------------------------- SwiGLU -----
// grid (FFN/256, S), block 256
__global__ void k_swiglu_bf16(const float* __restrict__ g,
                              unsigned short* __restrict__ u) {
  unsigned c = blockIdx.x * 256 + threadIdx.x;
  unsigned row = blockIdx.y;
  float g1 = g[row * (2u * FFN) + c];
  float g2 = g[row * (2u * FFN) + FFN + c];
  float sw = g1 / (1.f + __expf(-g1));
  u[row * (unsigned)FFN + c] = f2bf(sw * g2);
}

// --------------------------------------------------------------- driver ------
extern "C" void kernel_launch(void* const* d_in, const int* in_sizes, int n_in,
                              void* d_out, int out_size, void* d_ws, size_t ws_size,
                              hipStream_t stream) {
  const float* emb        = (const float*)d_in[0];
  // d_in[1] = attention_mask (causal, handled analytically)
  const float* Wq         = (const float*)d_in[2];
  const float* Wk         = (const float*)d_in[3];
  const float* Wv         = (const float*)d_in[4];
  const float* Wo         = (const float*)d_in[5];
  const float* attn_scale = (const float*)d_in[6];
  const float* ffn_scale  = (const float*)d_in[7];
  const float* W1         = (const float*)d_in[8];
  const float* W2         = (const float*)d_in[9];
  float* out = (float*)d_out;

  char* ws = (char*)d_ws;
  size_t off = 0;
  auto alloc = [&](size_t bytes) {
    char* p = ws + off;
    off += (bytes + 255) & ~(size_t)255;
    return p;
  };
  unsigned short* xn  = (unsigned short*)alloc((size_t)S_LEN * D_MODEL * 2);
  float*          qf  = (float*)         alloc((size_t)S_LEN * D_MODEL * 4);
  float*          kf  = (float*)         alloc((size_t)S_LEN * KVH * HD * 4);
  float*          vf  = (float*)         alloc((size_t)S_LEN * KVH * HD * 4);
  unsigned short* qbb = (unsigned short*)alloc((size_t)S_LEN * D_MODEL * 2);
  unsigned short* kbb = (unsigned short*)alloc((size_t)S_LEN * KVH * HD * 2);
  unsigned short* vbb = (unsigned short*)alloc((size_t)S_LEN * KVH * HD * 2);
  unsigned short* obb = (unsigned short*)alloc((size_t)S_LEN * D_MODEL * 2);
  float*          hbuf= (float*)         alloc((size_t)S_LEN * D_MODEL * 4);
  unsigned short* xn2 = (unsigned short*)alloc((size_t)S_LEN * D_MODEL * 2);
  float*          g   = (float*)         alloc((size_t)S_LEN * 2 * FFN * 4);
  unsigned short* u   = (unsigned short*)alloc((size_t)S_LEN * FFN * 2);

  // attn pre-norm
  k_rmsnorm_bf16<<<S_LEN, 256, 0, stream>>>(emb, attn_scale, xn);
  // Q/K/V projections (M tile = 128)
  k_gemm_bf16<<<dim3(D_MODEL / 128, S_LEN / 128), 256, 0, stream>>>(
      xn, Wq, qf, nullptr, S_LEN, D_MODEL, D_MODEL);
  k_gemm_bf16<<<dim3((KVH * HD) / 128, S_LEN / 128), 256, 0, stream>>>(
      xn, Wk, kf, nullptr, S_LEN, KVH * HD, D_MODEL);
  k_gemm_bf16<<<dim3((KVH * HD) / 128, S_LEN / 128), 256, 0, stream>>>(
      xn, Wv, vf, nullptr, S_LEN, KVH * HD, D_MODEL);
  // RoPE + bf16 conversion
  k_rope_bf16<<<dim3((NH * 64) / 256, S_LEN), 256, 0, stream>>>(qf, qbb, NH);
  k_rope_bf16<<<dim3(1, S_LEN), 256, 0, stream>>>(kf, kbb, KVH);
  k_f32_to_bf16<<<(S_LEN * KVH * HD + 255) / 256, 256, 0, stream>>>(
      vf, vbb, S_LEN * KVH * HD);
  // flash attention
  k_flash_attn<<<dim3(S_LEN / 64, NH), 128, 0, stream>>>(qbb, kbb, vbb, obb);
  // O projection + residual
  k_gemm_bf16<<<dim3(D_MODEL / 128, S_LEN / 128), 256, 0, stream>>>(
      obb, Wo, hbuf, emb, S_LEN, D_MODEL, D_MODEL);
  // ffn pre-norm
  k_rmsnorm_bf16<<<S_LEN, 256, 0, stream>>>(hbuf, ffn_scale, xn2);
  // W1 (GLU, doubled)
  k_gemm_bf16<<<dim3((2 * FFN) / 128, S_LEN / 128), 256, 0, stream>>>(
      xn2, W1, g, nullptr, S_LEN, 2 * FFN, D_MODEL);
  // SwiGLU
  k_swiglu_bf16<<<dim3(FFN / 256, S_LEN), 256, 0, stream>>>(g, u);
  // W2 + residual -> output
  k_gemm_bf16<<<dim3(D_MODEL / 128, S_LEN / 128), 256, 0, stream>>>(
      u, W2, out, hbuf, S_LEN, D_MODEL, FFN);
}
